// DeepLSTM_88081189307004
// MI455X (gfx1250) — matrix operand order repaired and checked by reading, not compiled, not run on hardware
//
#include <hip/hip_runtime.h>

// Problem dims
#define TSEQ 256
#define BATCH 64
#define IDIM 256
#define HDIM 1024
#define ODIM 256
#define GDIM 4096           // 4*HDIM
#define K1TOT (IDIM + HDIM)          // 1280, layer-1 K
#define K2TOT (IDIM + HDIM + HDIM)   // 2304, layer-2 K

typedef __attribute__((ext_vector_type(16))) __bf16 v16bf;
typedef __attribute__((ext_vector_type(8)))  __bf16 v8bf;
typedef __attribute__((ext_vector_type(8)))  float  v8f;

// ---------------------------------------------------------------------------
// Fragment load for V_WMMA_F32_16X16X32_BF16 (wave32) from a pre-offset
// per-lane pointer.
// A (16x32 MxK): lane<16 -> M=lane, K=0..15; lane>=16 ->
// M=lane-16, K=16..31  => two contiguous 16B loads.
// ---------------------------------------------------------------------------
__device__ __forceinline__ v16bf load_A_at(const __bf16* __restrict__ p) {
  v16bf out;
  *(v8bf*)(&out)       = *(const v8bf*)(p);
  *((v8bf*)(&out) + 1) = *(const v8bf*)(p + 8);
  return out;
}

// ---------------------------------------------------------------------------
// One K segment of the GEMM: NK k-tiles of 32, fixed A/B base pointers.
// pa: per-lane A pointer  = src + (row0+lane%16)*stride + (lane/16)*16
// pb: per-lane B pointer  = W   + (n0 +lane%16)*K      + (lane/16)*16
// B (32x16 KxN): lane<16 -> N=lane, K=0..15; lane>=16 -> N=lane-16, K=16..31
//   (one contiguous 32B load per lane from row-major W[N,K]).
// ---------------------------------------------------------------------------
template <int NK>
__device__ __forceinline__ void mma_seg(v8f& acc,
                                        const __bf16* __restrict__ pa, size_t step_a,
                                        const __bf16* __restrict__ pb) {
#pragma unroll 4
  for (int kt = 0; kt < NK; ++kt) {
    v16bf a = load_A_at(pa);
    v16bf b = *(const v16bf*)pb;
    // speculative prefetch of the B stream 512B ahead (L2 -> WGP$)
    __builtin_prefetch((const void*)(pb + 32 * 8), 0, 3);
    acc = __builtin_amdgcn_wmma_f32_16x16x32_bf16(false, a, false, b,
                                                  (short)0, acc, false, false);
    pa += step_a;   // 32 elements along K
    pb += 32;
  }
}

__device__ __forceinline__ float sigm(float x) {
  return 1.0f / (1.0f + __expf(-x));
}

// ---------------------------------------------------------------------------
// One LSTM layer for one timestep.
//   gates = [x_t | (haux) | hin] @ W^T + bias      (W row-major [4096,K] bf16)
//   fused cell update -> hout (bf16), c (f32 in place)
// grid: 64 blocks (16-wide hidden slice j0), 512 threads = 16 waves.
// wave -> (gate = w>>2, m-tile = w&3); each wave: one 16x16 tile, full K.
// ---------------------------------------------------------------------------
template <bool HAS_AUX>
__global__ void __launch_bounds__(512)
lstm_layer_kernel(const __bf16* __restrict__ xbf,   // [B,T,I] bf16
                  const __bf16* __restrict__ hin,   // [B,H] prev h of this layer
                  const __bf16* __restrict__ haux,  // layer2: current h1
                  const __bf16* __restrict__ W,     // [4096,K] bf16 row-major
                  const float*  __restrict__ bias,  // [4096] (b_ih + b_hh)
                  float*        __restrict__ c,     // [B,H] f32, in place
                  __bf16*       __restrict__ hout,  // [B,H] bf16
                  int t)
{
  constexpr int K = HAS_AUX ? K2TOT : K1TOT;
  __shared__ float g_lds[4][BATCH][17];   // +1 pad: lane vs lane+16 bank clash

  const int lane  = threadIdx.x & 31;
  const int wave  = threadIdx.x >> 5;
  const int gate  = wave >> 2;            // 0..3 (i,f,g,o)
  const int m0    = (wave & 3) << 4;      // batch tile
  const int j0    = blockIdx.x << 4;      // hidden slice
  const int n0    = (gate << 10) + j0;    // weight-row / gate-column base
  const int lrow  = lane & 15;
  const int lhalf = lane >> 4;

  // Bias folded into accumulator: D element col = lane&15 for all 8 rows.
  const float bv = bias[n0 + lrow];
  v8f acc = {bv, bv, bv, bv, bv, bv, bv, bv};

  // Per-lane base pointers for each K segment (computed once).
  const __bf16* pa_x = xbf + (size_t)(m0 + lrow) * (TSEQ * IDIM)
                           + (size_t)t * IDIM + (lhalf << 4);
  const __bf16* pa_h1 = (HAS_AUX ? haux : hin)
                        + (size_t)(m0 + lrow) * HDIM + (lhalf << 4);
  const __bf16* pb = W + (size_t)(n0 + lrow) * K + (lhalf << 4);

  mma_seg<IDIM / 32>(acc, pa_x, 32, pb);                       // x_t
  mma_seg<HDIM / 32>(acc, pa_h1, 32, pb + IDIM);               // h1
  if constexpr (HAS_AUX) {                                     // layer2: h2
    const __bf16* pa_h2 = hin + (size_t)(m0 + lrow) * HDIM + (lhalf << 4);
    mma_seg<HDIM / 32>(acc, pa_h2, 32, pb + IDIM + HDIM);
  }

  // D layout: element r -> row M = m0 + half*8 + r, col N = lane&15
  const int rbase = m0 + (lhalf << 3);
#pragma unroll
  for (int r = 0; r < 8; ++r)
    g_lds[gate][rbase + r][lrow] = acc[r];

  __syncthreads();

  // Fused cell update: 64 rows x 16 cols of this hidden slice, 2 per thread.
  for (int e = threadIdx.x; e < BATCH * 16; e += 512) {
    const int row = e >> 4;
    const int col = e & 15;
    const int j   = j0 + col;
    const float gi = g_lds[0][row][col];
    const float gf = g_lds[1][row][col];
    const float gg = g_lds[2][row][col];
    const float go = g_lds[3][row][col];
    const float si = sigm(gi);
    const float sf = sigm(gf);
    const float so = sigm(go);
    const float tg = tanhf(gg);
    const int idx  = row * HDIM + j;
    const float cn = sf * c[idx] + si * tg;
    c[idx]    = cn;
    hout[idx] = (__bf16)(so * tanhf(cn));
  }
}

// ---------------------------------------------------------------------------
// y_t = [x_t | h2_t] @ W_l^T + b_l   -> out[B,T,O] f32
// grid: 16 blocks (N tiles), 128 threads = 4 waves (M tiles), 1 tile/wave.
// ---------------------------------------------------------------------------
__global__ void __launch_bounds__(128)
linear_kernel(const __bf16* __restrict__ xbf,
              const __bf16* __restrict__ h2,    // [B,H] current h2
              const __bf16* __restrict__ Wl,    // [256,1280] bf16
              const float*  __restrict__ bl,    // [256] f32
              float*        __restrict__ out,   // [B,T,O] f32
              int t)
{
  const int lane  = threadIdx.x & 31;
  const int m0    = (threadIdx.x >> 5) << 4;
  const int n0    = blockIdx.x << 4;
  const int lrow  = lane & 15;
  const int lhalf = lane >> 4;

  const float bv = bl[n0 + lrow];
  v8f acc = {bv, bv, bv, bv, bv, bv, bv, bv};

  const __bf16* pa_x = xbf + (size_t)(m0 + lrow) * (TSEQ * IDIM)
                           + (size_t)t * IDIM + (lhalf << 4);
  const __bf16* pa_h = h2 + (size_t)(m0 + lrow) * HDIM + (lhalf << 4);
  const __bf16* pb   = Wl + (size_t)(n0 + lrow) * K1TOT + (lhalf << 4);

  mma_seg<IDIM / 32>(acc, pa_x, 32, pb);
  mma_seg<HDIM / 32>(acc, pa_h, 32, pb + IDIM);

  const int b0  = m0 + (lhalf << 3);
  const int col = n0 + lrow;
#pragma unroll
  for (int r = 0; r < 8; ++r)
    out[(size_t)(b0 + r) * TSEQ * ODIM + (size_t)t * ODIM + col] = acc[r];
}

// --------------------------- setup / prep kernels ---------------------------
__global__ void cvt_f32_bf16(const float* __restrict__ in,
                             __bf16* __restrict__ outp, int n) {
  int i = blockIdx.x * blockDim.x + threadIdx.x;
  if (i < n) outp[i] = (__bf16)in[i];
}

// out[n, 0:K1] = A[n, :] ; out[n, K1:K1+K2] = Bm[n, :]   (f32 -> bf16)
__global__ void concat_w(const float* __restrict__ A, const float* __restrict__ Bm,
                         __bf16* __restrict__ outp, int K1, int K2, long total) {
  long i = (long)blockIdx.x * blockDim.x + threadIdx.x;
  if (i >= total) return;
  const int Kt = K1 + K2;
  const long nrow = i / Kt;
  const int  k    = (int)(i - nrow * Kt);
  const float v = (k < K1) ? A[nrow * K1 + k] : Bm[nrow * K2 + (k - K1)];
  outp[i] = (__bf16)v;
}

__global__ void vec_add(const float* a, const float* b, float* o, int n) {
  int i = blockIdx.x * blockDim.x + threadIdx.x;
  if (i < n) o[i] = a[i] + b[i];
}
__global__ void zero_f32(float* p, int n) {
  int i = blockIdx.x * blockDim.x + threadIdx.x;
  if (i < n) p[i] = 0.0f;
}
__global__ void zero_bf16(__bf16* p, int n) {
  int i = blockIdx.x * blockDim.x + threadIdx.x;
  if (i < n) p[i] = (__bf16)0.0f;
}

// ---------------------------------------------------------------------------
extern "C" void kernel_launch(void* const* d_in, const int* in_sizes, int n_in,
                              void* d_out, int out_size, void* d_ws, size_t ws_size,
                              hipStream_t stream) {
  (void)in_sizes; (void)n_in; (void)out_size; (void)ws_size;

  const float* x     = (const float*)d_in[0];
  const float* W_ih1 = (const float*)d_in[1];
  const float* W_hh1 = (const float*)d_in[2];
  const float* b_ih1 = (const float*)d_in[3];
  const float* b_hh1 = (const float*)d_in[4];
  const float* W_ih2 = (const float*)d_in[5];
  const float* W_hh2 = (const float*)d_in[6];
  const float* b_ih2 = (const float*)d_in[7];
  const float* b_hh2 = (const float*)d_in[8];
  const float* W_l   = (const float*)d_in[9];
  const float* b_l   = (const float*)d_in[10];

  // Workspace layout (256B aligned), ~40 MB total.
  char* ws = (char*)d_ws;
  size_t off = 0;
  auto alloc = [&](size_t bytes) -> void* {
    off = (off + 255) & ~(size_t)255;
    void* p = ws + off;
    off += bytes;
    return p;
  };
  __bf16* xbf = (__bf16*)alloc((size_t)BATCH * TSEQ * IDIM * 2);   // 8 MB
  __bf16* Wg1 = (__bf16*)alloc((size_t)GDIM * K1TOT * 2);          // 10 MB
  __bf16* Wg2 = (__bf16*)alloc((size_t)GDIM * K2TOT * 2);          // 18 MB
  __bf16* Wlc = (__bf16*)alloc((size_t)ODIM * K1TOT * 2);          // 0.6 MB
  float*  b1  = (float*)alloc((size_t)GDIM * 4);
  float*  b2  = (float*)alloc((size_t)GDIM * 4);
  __bf16* h1b = (__bf16*)alloc((size_t)2 * BATCH * HDIM * 2);      // ping-pong
  __bf16* h2b = (__bf16*)alloc((size_t)2 * BATCH * HDIM * 2);
  float*  c1  = (float*)alloc((size_t)BATCH * HDIM * 4);
  float*  c2  = (float*)alloc((size_t)BATCH * HDIM * 4);

  // ---- one-time-per-call prep (ws is poisoned; must re-init every call) ----
  {
    int n = BATCH * TSEQ * IDIM;
    cvt_f32_bf16<<<(n + 255) / 256, 256, 0, stream>>>(x, xbf, n);
    long t1 = (long)GDIM * K1TOT;
    concat_w<<<(unsigned)((t1 + 255) / 256), 256, 0, stream>>>(W_ih1, W_hh1, Wg1,
                                                               IDIM, HDIM, t1);
    long t2 = (long)GDIM * K2TOT;
    concat_w<<<(unsigned)((t2 + 255) / 256), 256, 0, stream>>>(W_ih2, W_hh2, Wg2,
                                                               K1TOT, HDIM, t2);
    n = ODIM * K1TOT;
    cvt_f32_bf16<<<(n + 255) / 256, 256, 0, stream>>>(W_l, Wlc, n);
    vec_add<<<(GDIM + 255) / 256, 256, 0, stream>>>(b_ih1, b_hh1, b1, GDIM);
    vec_add<<<(GDIM + 255) / 256, 256, 0, stream>>>(b_ih2, b_hh2, b2, GDIM);
    n = BATCH * HDIM;
    zero_bf16<<<(n + 255) / 256, 256, 0, stream>>>(h1b, n);   // ping 0 only
    zero_bf16<<<(n + 255) / 256, 256, 0, stream>>>(h2b, n);
    zero_f32 <<<(n + 255) / 256, 256, 0, stream>>>(c1, n);
    zero_f32 <<<(n + 255) / 256, 256, 0, stream>>>(c2, n);
  }

  // ---- sequential recurrence over T ----
  const size_t HS = (size_t)BATCH * HDIM;
  for (int t = 0; t < TSEQ; ++t) {
    const __bf16* h1in  = h1b + (size_t)(t & 1) * HS;
    __bf16*       h1out = h1b + (size_t)((t + 1) & 1) * HS;
    const __bf16* h2in  = h2b + (size_t)(t & 1) * HS;
    __bf16*       h2out = h2b + (size_t)((t + 1) & 1) * HS;

    lstm_layer_kernel<false><<<HDIM / 16, 512, 0, stream>>>(
        xbf, h1in, nullptr, Wg1, b1, c1, h1out, t);
    lstm_layer_kernel<true><<<HDIM / 16, 512, 0, stream>>>(
        xbf, h2in, h1out, Wg2, b2, c2, h2out, t);
    linear_kernel<<<ODIM / 16, 128, 0, stream>>>(
        xbf, h2out, Wlc, b_l, (float*)d_out, t);
  }
}
